// HeteroGATLayerLossyReal_7318624272784
// MI455X (gfx1250) — matrix operand order repaired and checked by reading, not compiled, not run on hardware
//
#include <hip/hip_runtime.h>

// MI455X / gfx1250 HeteroGAT forward.
// Phases (all on one stream, so buffer reuse is ordering-safe):
//   1) 3 self GEMMs (bf16 WMMA)  -> Wh_self[t]            (ws)
//   2) 3 fused dot kernels       -> sdst[r][n][h] (9xNx4) (ws)
//   3) zero d_out
//   4) per relation: GEMM -> Wh_rel (aliases Wh_self), then fused
//      gather + attention + softmax(4 heads) + atomic scatter into d_out
//   5) in-place ReLU on d_out
// Workspace requirement: 9*N*4 + 3*N*64 floats ~= 92 MB.

typedef __attribute__((ext_vector_type(16))) __bf16 v16bf;
typedef __attribute__((ext_vector_type(8)))  float  v8f;

// ---------------------------------------------------------------------------
// GEMM: Y[nrows,64] = X[nrows,64] @ W[64,64] + bias, fp32 in/out, bf16 WMMA.
// One wave computes a 16-row x 64-col strip: 4 n-tiles x 2 k-steps = 8 WMMA.
// W is staged in LDS pre-swizzled into per-lane 32B B-fragments.
// Fast-path epilogue: when the whole 16-row tile is in range (always true
// for N % 16 == 0), stores use one base pointer + immediate offsets, no
// per-element EXEC juggling.
// ---------------------------------------------------------------------------
__global__ __launch_bounds__(256) void gemm64_wmma_bf16(
    const float* __restrict__ X, const float* __restrict__ W,
    const float* __restrict__ bias, float* __restrict__ Y, int nrows)
{
  __shared__ __bf16 sW[4096];   // [kstep(2)][ntile(4)][lane(32)][e(16)]
  __shared__ float  sB[64];
  const int tid = threadIdx.x;

  // Cooperative load + f32->bf16 convert + swizzle of W (row-major W[k*64+n]).
  // B-fragment layout (32x16 bf16): lane = (n&15) + ((k&8) ? 16:0),
  // element e = (k&7) + ((k%32>=16) ? 8:0) (same K order as A fragments).
  for (int i = tid; i < 4096; i += 256) {
    const int k = i >> 6, n = i & 63;
    const int kstep = k >> 5, kk = k & 31;
    const int lane  = (n & 15) + ((kk & 8) << 1);
    const int e     = (kk & 7) | ((kk & 16) >> 1);
    const int frag  = (kstep << 2) | (n >> 4);      // 0..7
    sW[(((frag << 5) | lane) << 4) + e] = (__bf16)W[i];
  }
  if (tid < 64) sB[tid] = bias[tid];
  __syncthreads();

  const int lane = tid & 31;
  const int wave = tid >> 5;
  const int rowbase = (blockIdx.x * 8 + wave) << 4;
  if (rowbase >= nrows) return;

  const int m = lane & 15;
  int row = rowbase + m;
  if (row >= nrows) row = nrows - 1;               // clamp; stores are guarded
  const float4* xv = (const float4*)(X + (size_t)row * 64);
  const int klo = (lane & 16) ? 8 : 0;             // A K-half per lane group

  // A fragments (16x32 bf16 each): lanes<16 K{0..7,16..23}, lanes>=16 K{8..15,24..31}
  v16bf a[2];
  #pragma unroll
  for (int s = 0; s < 2; ++s) {
    const int kb = s << 5;
    const float4 p0 = xv[(kb + klo) >> 2];
    const float4 p1 = xv[((kb + klo) >> 2) + 1];
    const float4 q0 = xv[(kb + 16 + klo) >> 2];
    const float4 q1 = xv[((kb + 16 + klo) >> 2) + 1];
    a[s][0]=(__bf16)p0.x; a[s][1]=(__bf16)p0.y; a[s][2]=(__bf16)p0.z; a[s][3]=(__bf16)p0.w;
    a[s][4]=(__bf16)p1.x; a[s][5]=(__bf16)p1.y; a[s][6]=(__bf16)p1.z; a[s][7]=(__bf16)p1.w;
    a[s][8]=(__bf16)q0.x; a[s][9]=(__bf16)q0.y; a[s][10]=(__bf16)q0.z; a[s][11]=(__bf16)q0.w;
    a[s][12]=(__bf16)q1.x; a[s][13]=(__bf16)q1.y; a[s][14]=(__bf16)q1.z; a[s][15]=(__bf16)q1.w;
  }

  const int rmod = (lane & 16) ? 8 : 0;            // C/D row half per lane group
  const bool full = (rowbase + 16) <= nrows;       // wave-uniform
  float* ybase = Y + (size_t)(rowbase + rmod) * 64 + m;

  #pragma unroll
  for (int t = 0; t < 4; ++t) {
    const float bv = sB[(t << 4) | m];             // bias depends only on column
    v8f c;
    #pragma unroll
    for (int v = 0; v < 8; ++v) c[v] = bv;
    #pragma unroll
    for (int s = 0; s < 2; ++s) {
      const int frag = (s << 2) | t;
      const v16bf* bw = (const v16bf*)&sW[((frag << 5) | lane) << 4];
      c = __builtin_amdgcn_wmma_f32_16x16x32_bf16(
              /*neg_a=*/false, a[s], /*neg_b=*/false, *bw,
              /*c_mod=*/(short)0, c, /*reuse_a=*/false, /*reuse_b=*/false);
    }
    // store: element v -> row rowbase + rmod + v, col t*16 + m
    if (full) {
      float* yp = ybase + (t << 4);
      #pragma unroll
      for (int v = 0; v < 8; ++v) yp[(size_t)v * 64] = c[v];
    } else {
      #pragma unroll
      for (int v = 0; v < 8; ++v) {
        const int rr = rowbase + rmod + v;
        if (rr < nrows) Y[(size_t)rr * 64 + (t << 4) + m] = c[v];
      }
    }
  }
}

// ---------------------------------------------------------------------------
// sdst[rX][i][h] = sum_d Wh_self[i,h,d] * adstX[h,d]  for the 3 relations
// whose destination is this node type (reads Wh_self once for all three).
// ---------------------------------------------------------------------------
__global__ __launch_bounds__(256) void sdst3_kernel(
    const float* __restrict__ WhSelf,
    const float* __restrict__ a0, const float* __restrict__ a1,
    const float* __restrict__ a2,
    float* __restrict__ s0, float* __restrict__ s1, float* __restrict__ s2,
    int n)
{
  __shared__ float4 sa[3][16];
  if (threadIdx.x < 16)      sa[0][threadIdx.x]      = ((const float4*)a0)[threadIdx.x];
  else if (threadIdx.x < 32) sa[1][threadIdx.x - 16] = ((const float4*)a1)[threadIdx.x - 16];
  else if (threadIdx.x < 48) sa[2][threadIdx.x - 32] = ((const float4*)a2)[threadIdx.x - 32];
  __syncthreads();

  const int i = blockIdx.x * blockDim.x + threadIdx.x;
  if (i >= n) return;
  const float4* w = (const float4*)(WhSelf + (size_t)i * 64);
  float r0[4] = {0,0,0,0}, r1[4] = {0,0,0,0}, r2[4] = {0,0,0,0};
  #pragma unroll
  for (int c = 0; c < 16; ++c) {
    const float4 wv = w[c];
    const int h = c >> 2;
    const float4 v0 = sa[0][c], v1 = sa[1][c], v2 = sa[2][c];
    r0[h] += wv.x*v0.x + wv.y*v0.y + wv.z*v0.z + wv.w*v0.w;
    r1[h] += wv.x*v1.x + wv.y*v1.y + wv.z*v1.z + wv.w*v1.w;
    r2[h] += wv.x*v2.x + wv.y*v2.y + wv.z*v2.z + wv.w*v2.w;
  }
  ((float4*)s0)[i] = make_float4(r0[0], r0[1], r0[2], r0[3]);
  ((float4*)s1)[i] = make_float4(r1[0], r1[1], r1[2], r1[3]);
  ((float4*)s2)[i] = make_float4(r2[0], r2[1], r2[2], r2[3]);
}

// ---------------------------------------------------------------------------
// Edge phase: gather Wh[src], logits = lrelu(dot(whe,asrc)+sdst[dst]),
// softmax over the 4 heads, atomic scatter whe*alpha into out[dst].
// ---------------------------------------------------------------------------
__global__ __launch_bounds__(256) void edge_kernel(
    const float* __restrict__ Wh, const int* __restrict__ src,
    const int* __restrict__ dst, const float* __restrict__ asrc,
    const float* __restrict__ sdst, float* __restrict__ out, int nedges)
{
  __shared__ float4 sa[16];
  if (threadIdx.x < 16) sa[threadIdx.x] = ((const float4*)asrc)[threadIdx.x];
  __syncthreads();

  const int e = blockIdx.x * blockDim.x + threadIdx.x;
  if (e >= nedges) return;
  const int s = src[e], d = dst[e];

  const float4* w = (const float4*)(Wh + (size_t)s * 64);
  float4 wv[16];
  float logit[4] = {0.f, 0.f, 0.f, 0.f};
  #pragma unroll
  for (int c = 0; c < 16; ++c) {
    wv[c] = w[c];
    const float4 av = sa[c];
    logit[c >> 2] += wv[c].x*av.x + wv[c].y*av.y + wv[c].z*av.z + wv[c].w*av.w;
  }
  const float4 sd = ((const float4*)sdst)[d];
  logit[0] += sd.x; logit[1] += sd.y; logit[2] += sd.z; logit[3] += sd.w;
  #pragma unroll
  for (int h = 0; h < 4; ++h)
    logit[h] = (logit[h] >= 0.f) ? logit[h] : 0.2f * logit[h];

  const float mx = fmaxf(fmaxf(logit[0], logit[1]), fmaxf(logit[2], logit[3]));
  float al[4], sum = 0.f;
  #pragma unroll
  for (int h = 0; h < 4; ++h) { al[h] = __expf(logit[h] - mx); sum += al[h]; }
  const float inv = 1.f / sum;

  float* o = out + (size_t)d * 64;
  #pragma unroll
  for (int c = 0; c < 16; ++c) {
    const float a = al[c >> 2] * inv;
    unsafeAtomicAdd(&o[4*c + 0], wv[c].x * a);   // global_atomic_add_f32
    unsafeAtomicAdd(&o[4*c + 1], wv[c].y * a);
    unsafeAtomicAdd(&o[4*c + 2], wv[c].z * a);
    unsafeAtomicAdd(&o[4*c + 3], wv[c].w * a);
  }
}

__global__ void zero_kernel(float* __restrict__ p, size_t n4) {
  size_t i = (size_t)blockIdx.x * blockDim.x + threadIdx.x;
  const size_t stride = (size_t)gridDim.x * blockDim.x;
  float4 z = make_float4(0.f, 0.f, 0.f, 0.f);
  for (; i < n4; i += stride) ((float4*)p)[i] = z;
}

__global__ void relu_kernel(float* __restrict__ p, size_t n4) {
  size_t i = (size_t)blockIdx.x * blockDim.x + threadIdx.x;
  const size_t stride = (size_t)gridDim.x * blockDim.x;
  for (; i < n4; i += stride) {
    float4 v = ((float4*)p)[i];
    v.x = fmaxf(v.x, 0.f); v.y = fmaxf(v.y, 0.f);
    v.z = fmaxf(v.z, 0.f); v.w = fmaxf(v.w, 0.f);
    ((float4*)p)[i] = v;
  }
}

extern "C" void kernel_launch(void* const* d_in, const int* in_sizes, int n_in,
                              void* d_out, int out_size, void* d_ws, size_t ws_size,
                              hipStream_t stream)
{
  const int N = in_sizes[0] / 64;     // nodes per type
  const int E = in_sizes[9 + 4];      // edges per relation (src_c1c1)

  const float *feat[3], *Wself[3], *bself[3];
  for (int t = 0; t < 3; ++t) {
    feat[t]  = (const float*)d_in[3*t + 0];
    Wself[t] = (const float*)d_in[3*t + 1];
    bself[t] = (const float*)d_in[3*t + 2];
  }
  const float *Wr[9], *br[9], *asrc[9], *adst[9];
  const int *esrc[9], *edst[9];
  for (int r = 0; r < 9; ++r) {
    const int b = 9 + 6*r;
    Wr[r]   = (const float*)d_in[b + 0];
    br[r]   = (const float*)d_in[b + 1];
    asrc[r] = (const float*)d_in[b + 2];
    adst[r] = (const float*)d_in[b + 3];
    esrc[r] = (const int*)d_in[b + 4];
    edst[r] = (const int*)d_in[b + 5];
  }

  float* out     = (float*)d_out;
  float* sdstAll = (float*)d_ws;                        // 9 * N * 4 floats
  float* whself  = sdstAll + (size_t)9 * N * 4;         // 3 * N * 64 floats
  float* whrel   = whself;  // alias: safe, stream-ordered after sdst kernels

  const dim3 blk(256);
  const int gemmGrid = (N + 127) / 128;                 // 8 waves x 16 rows
  const int nGrid    = (N + 255) / 256;
  const int eGrid    = (E + 255) / 256;

  // 1) self projections
  for (int t = 0; t < 3; ++t)
    gemm64_wmma_bf16<<<gemmGrid, blk, 0, stream>>>(
        feat[t], Wself[t], bself[t], whself + (size_t)t * N * 64, N);

  // 2) per-node dst attention scalars (relations with dst==t are r = t, 3+t, 6+t)
  for (int t = 0; t < 3; ++t)
    sdst3_kernel<<<nGrid, blk, 0, stream>>>(
        whself + (size_t)t * N * 64,
        adst[t], adst[3 + t], adst[6 + t],
        sdstAll + (size_t)(t)     * N * 4,
        sdstAll + (size_t)(3 + t) * N * 4,
        sdstAll + (size_t)(6 + t) * N * 4, N);

  // 3) zero the (poisoned) output accumulator
  const size_t out4 = (size_t)3 * N * 64 / 4;
  zero_kernel<<<2048, blk, 0, stream>>>(out, out4);

  // 4) relations: projection GEMM then fused edge kernel
  for (int r = 0; r < 9; ++r) {
    const int s = r / 3, d = r % 3;
    gemm64_wmma_bf16<<<gemmGrid, blk, 0, stream>>>(feat[s], Wr[r], br[r], whrel, N);
    edge_kernel<<<eGrid, blk, 0, stream>>>(
        whrel, esrc[r], edst[r], asrc[r],
        sdstAll + (size_t)r * N * 4, out + (size_t)d * N * 64, E);
  }

  // 5) final ReLU in place
  relu_kernel<<<2048, blk, 0, stream>>>(out, out4);
}
